// PAE_25640954757479
// MI455X (gfx1250) — matrix operand order repaired and checked by reading, not compiled, not run on hardware
//
#include <hip/hip_runtime.h>

#define N_NODES_C 100000
#define N_EDGES_C 1000000

typedef __bf16 bf16_t;
typedef __attribute__((ext_vector_type(16))) __bf16 v16bf;
typedef __attribute__((ext_vector_type(8)))  __bf16 v8bf;
typedef __attribute__((ext_vector_type(8)))  float  v8f;
typedef __attribute__((ext_vector_type(4)))  float  v4f;

// Load a v16bf WMMA fragment from two 16B-aligned LDS chunks of 8 bf16 each.
__device__ __forceinline__ v16bf pae_ld2x8(const bf16_t* p0, const bf16_t* p1) {
  v8bf lo = *(const v8bf*)p0;
  v8bf hi = *(const v8bf*)p1;
  v16bf r;
#pragma unroll
  for (int i = 0; i < 8; ++i) { r[i] = lo[i]; r[i + 8] = hi[i]; }
  return r;
}

// ---------------- zero counts + stats ----------------
__global__ __launch_bounds__(256) void pae_zero(int* __restrict__ counts,
                                                float* __restrict__ stats) {
  int i = blockIdx.x * 256 + threadIdx.x;
  if (i < 2 * N_NODES_C) counts[i] = 0;
  if (i < 512) stats[i] = 0.0f;
}

// ---------------- per-node degree counts (one per side) ----------------
__global__ __launch_bounds__(256) void pae_count(const int* __restrict__ ei,
                                                 int* __restrict__ c1,
                                                 int* __restrict__ c2) {
  int i = blockIdx.x * 256 + threadIdx.x;
  if (i < N_EDGES_C) {
    atomicAdd(&c1[ei[i]], 1);
    atomicAdd(&c2[ei[N_EDGES_C + i]], 1);
  }
}

// ---------------- GEMM1: H = relu(x @ W1 + b1), bf16 out ----------------
__global__ __launch_bounds__(256) void pae_gemm1(const float* __restrict__ x,
                                                 const float* __restrict__ W1,
                                                 const float* __restrict__ b1,
                                                 bf16_t* __restrict__ H,
                                                 int nNodes) {
  __shared__ bf16_t Xs[64 * 80];    // 64 rows x 64 k (stride 80 -> 160B, 16B aligned)
  __shared__ bf16_t W1t[128 * 72];  // transposed: [col][k], stride 72 -> 144B (16B mult)
  const int tid = threadIdx.x;
  const int base = blockIdx.x * 64;
  const bool full = (base + 64 <= nNodes);  // block-uniform fast path

  if (full) {
    for (int idx = tid; idx < 64 * 64; idx += 256) {
      int r = idx >> 6, c = idx & 63;
      // x is read exactly once across the whole pipeline: keep it out of L2
      Xs[r * 80 + c] = (bf16_t)__builtin_nontemporal_load(&x[(base + r) * 64 + c]);
    }
  } else {
    for (int idx = tid; idx < 64 * 64; idx += 256) {
      int r = idx >> 6, c = idx & 63;
      int row = base + r;
      float v = (row < nNodes) ? x[row * 64 + c] : 0.0f;
      Xs[r * 80 + c] = (bf16_t)v;
    }
  }
  for (int idx = tid; idx < 64 * 128; idx += 256) {
    int k = idx >> 7, c = idx & 127;
    W1t[c * 72 + k] = (bf16_t)W1[idx];
  }
  __syncthreads();

  const int lane = tid & 31, wave = tid >> 5;
  const int mt = wave >> 1, nh = wave & 1;       // 4 M-tiles x 2 N-halves
  const int l16 = lane & 15, h = lane >> 4;
  const int arow = mt * 16 + l16;

  v8f acc[4] = {};
#pragma unroll
  for (int kk = 0; kk < 64; kk += 32) {
    v16bf a = pae_ld2x8(&Xs[arow * 80 + kk + h * 8],
                        &Xs[arow * 80 + kk + 16 + h * 8]);
#pragma unroll
    for (int t = 0; t < 4; ++t) {
      int col = nh * 64 + t * 16 + l16;
      const bf16_t* bp = &W1t[col * 72 + kk + h * 16];
      v16bf b = pae_ld2x8(bp - h * 8, bp - h * 8 + 16);
      acc[t] = __builtin_amdgcn_wmma_f32_16x16x32_bf16(
          false, a, false, b, (short)0, acc[t], false, false);
    }
  }

  if (full) {
#pragma unroll
    for (int t = 0; t < 4; ++t) {
      int col = nh * 64 + t * 16 + l16;
      float bias = b1[col];
#pragma unroll
      for (int r = 0; r < 8; ++r) {
        int row = base + mt * 16 + h * 8 + r;
        H[row * 128 + col] = (bf16_t)fmaxf(acc[t][r] + bias, 0.0f);
      }
    }
  } else {
#pragma unroll
    for (int t = 0; t < 4; ++t) {
      int col = nh * 64 + t * 16 + l16;
      float bias = b1[col];
#pragma unroll
      for (int r = 0; r < 8; ++r) {
        int row = base + mt * 16 + h * 8 + r;
        if (row < nNodes)
          H[row * 128 + col] = (bf16_t)fmaxf(acc[t][r] + bias, 0.0f);
      }
    }
  }
}

// ------- count-weighted BN statistics: sum & sumsq per hidden, per side -------
__global__ __launch_bounds__(256) void pae_stats(const bf16_t* __restrict__ H,
                                                 const int* __restrict__ c1,
                                                 const int* __restrict__ c2,
                                                 float* __restrict__ stats,
                                                 int nNodes) {
  int col = threadIdx.x & 127;
  int rsub = threadIdx.x >> 7;  // 0..1
  int rowEnd = blockIdx.x * 128 + 128;
  if (rowEnd > nNodes) rowEnd = nNodes;
  float s1 = 0.f, q1 = 0.f, s2 = 0.f, q2 = 0.f;
  for (int r = blockIdx.x * 128 + rsub; r < rowEnd; r += 2) {
    float hv = (float)H[r * 128 + col];
    float w1 = (float)c1[r];
    float w2 = (float)c2[r];
    s1 += w1 * hv; q1 += w1 * hv * hv;
    s2 += w2 * hv; q2 += w2 * hv * hv;
  }
  atomicAdd(&stats[col], s1);
  atomicAdd(&stats[128 + col], q1);
  atomicAdd(&stats[256 + col], s2);
  atomicAdd(&stats[384 + col], q2);
}

// ---- fold BN into second linear: W2p_s = diag(scale_s) W2 (bf16), b2p_s ----
__global__ __launch_bounds__(256) void pae_finalize(const float* __restrict__ stats,
                                                    const float* __restrict__ gamma,
                                                    const float* __restrict__ beta,
                                                    const float* __restrict__ W2,
                                                    const float* __restrict__ b2,
                                                    bf16_t* __restrict__ W2p1,
                                                    bf16_t* __restrict__ W2p2,
                                                    float* __restrict__ b2p1,
                                                    float* __restrict__ b2p2) {
  __shared__ float scale1[128], shift1[128], scale2[128], shift2[128];
  int t = threadIdx.x;
  const float invE = 1.0f / (float)N_EDGES_C;
  if (t < 128) {
    float m1 = stats[t] * invE;
    float v1 = stats[128 + t] * invE - m1 * m1;
    float sc1 = gamma[t] * rsqrtf(v1 + 1e-5f);
    scale1[t] = sc1; shift1[t] = beta[t] - m1 * sc1;
    float m2 = stats[256 + t] * invE;
    float v2 = stats[384 + t] * invE - m2 * m2;
    float sc2 = gamma[t] * rsqrtf(v2 + 1e-5f);
    scale2[t] = sc2; shift2[t] = beta[t] - m2 * sc2;
  }
  __syncthreads();
  for (int idx = t; idx < 128 * 128; idx += 256) {
    int k = idx >> 7;
    float w = W2[idx];
    W2p1[idx] = (bf16_t)(scale1[k] * w);
    W2p2[idx] = (bf16_t)(scale2[k] * w);
  }
  if (t < 128) {
    float a1 = b2[t], a2 = b2[t];
    for (int k = 0; k < 128; ++k) {
      float w = W2[k * 128 + t];
      a1 += shift1[k] * w;
      a2 += shift2[k] * w;
    }
    b2p1[t] = a1; b2p2[t] = a2;
  }
}

// ---------------- GEMM2: G = H @ W2p + b2p (f32 out), launched per side ----------------
__global__ __launch_bounds__(256) void pae_gemm2(const bf16_t* __restrict__ H,
                                                 const bf16_t* __restrict__ W2p,
                                                 const float* __restrict__ b2p,
                                                 float* __restrict__ G,
                                                 int nNodes) {
  __shared__ bf16_t Hs[64 * 136];    // 64 rows x 128 k, stride 136 -> 272B (16B mult)
  __shared__ bf16_t Wt[128 * 136];   // transposed [col][k]
  const int tid = threadIdx.x;
  const int base = blockIdx.x * 64;
  const bool full = (base + 64 <= nNodes);

  if (full) {
    for (int idx = tid; idx < 64 * 128; idx += 256) {
      int r = idx >> 7, c = idx & 127;
      Hs[r * 136 + c] = H[(base + r) * 128 + c];
    }
  } else {
    for (int idx = tid; idx < 64 * 128; idx += 256) {
      int r = idx >> 7, c = idx & 127;
      int row = base + r;
      Hs[r * 136 + c] = (row < nNodes) ? H[row * 128 + c] : (bf16_t)0.0f;
    }
  }
  for (int idx = tid; idx < 128 * 128; idx += 256) {
    int k = idx >> 7, c = idx & 127;
    Wt[c * 136 + k] = W2p[idx];
  }
  __syncthreads();

  const int lane = tid & 31, wave = tid >> 5;
  const int mt = wave >> 1, nh = wave & 1;
  const int l16 = lane & 15, h = lane >> 4;
  const int arow = mt * 16 + l16;

  v8f acc[4] = {};
#pragma unroll
  for (int kk = 0; kk < 128; kk += 32) {
    v16bf a = pae_ld2x8(&Hs[arow * 136 + kk + h * 8],
                        &Hs[arow * 136 + kk + 16 + h * 8]);
#pragma unroll
    for (int t = 0; t < 4; ++t) {
      int col = nh * 64 + t * 16 + l16;
      const bf16_t* bp = &Wt[col * 136 + kk + h * 16];
      v16bf b = pae_ld2x8(bp - h * 8, bp - h * 8 + 16);
      acc[t] = __builtin_amdgcn_wmma_f32_16x16x32_bf16(
          false, a, false, b, (short)0, acc[t], false, false);
    }
  }

  if (full) {
#pragma unroll
    for (int t = 0; t < 4; ++t) {
      int col = nh * 64 + t * 16 + l16;
      float bias = b2p[col];
#pragma unroll
      for (int r = 0; r < 8; ++r) {
        int row = base + mt * 16 + h * 8 + r;
        G[row * 128 + col] = acc[t][r] + bias;
      }
    }
  } else {
#pragma unroll
    for (int t = 0; t < 4; ++t) {
      int col = nh * 64 + t * 16 + l16;
      float bias = b2p[col];
#pragma unroll
      for (int r = 0; r < 8; ++r) {
        int row = base + mt * 16 + h * 8 + r;
        if (row < nNodes) G[row * 128 + col] = acc[t][r] + bias;
      }
    }
  }
}

// ---------------- per-node L2 norms (one wave per node) ----------------
__global__ __launch_bounds__(256) void pae_norm(const float* __restrict__ G1,
                                                const float* __restrict__ G2,
                                                float* __restrict__ n1,
                                                float* __restrict__ n2,
                                                int nNodes) {
  int lane = threadIdx.x & 31;
  int node = (blockIdx.x * 256 + threadIdx.x) >> 5;
  if (node >= nNodes) return;
  v4f a = *(const v4f*)&G1[(size_t)node * 128 + lane * 4];
  float s = a[0] * a[0] + a[1] * a[1] + a[2] * a[2] + a[3] * a[3];
#pragma unroll
  for (int m = 16; m >= 1; m >>= 1) s += __shfl_xor(s, m, 32);
  if (lane == 0) n1[node] = sqrtf(s);
  v4f b = *(const v4f*)&G2[(size_t)node * 128 + lane * 4];
  float s2 = b[0] * b[0] + b[1] * b[1] + b[2] * b[2] + b[3] * b[3];
#pragma unroll
  for (int m = 16; m >= 1; m >>= 1) s2 += __shfl_xor(s2, m, 32);
  if (lane == 0) n2[node] = sqrtf(s2);
}

// ---------------- per-edge cosine: 16 lanes per edge, L2-resident gathers ----------------
__global__ __launch_bounds__(256) void pae_edge(const int* __restrict__ ei,
                                                const float* __restrict__ G1,
                                                const float* __restrict__ G2,
                                                const float* __restrict__ n1,
                                                const float* __restrict__ n2,
                                                float* __restrict__ out) {
  int tid = blockIdx.x * 256 + threadIdx.x;
  int l16 = threadIdx.x & 15;
  int e = tid >> 4;
  if (e >= N_EDGES_C) return;
  int src = ei[e];
  int dst = ei[N_EDGES_C + e];
  const float* p1 = &G1[(size_t)src * 128 + l16 * 8];
  const float* p2 = &G2[(size_t)dst * 128 + l16 * 8];
  v4f a0 = *(const v4f*)p1;
  v4f a1 = *(const v4f*)(p1 + 4);
  v4f b0 = *(const v4f*)p2;
  v4f b1 = *(const v4f*)(p2 + 4);
  float acc = 0.f;
#pragma unroll
  for (int i = 0; i < 4; ++i) acc = fmaf(a0[i], b0[i], acc);
#pragma unroll
  for (int i = 0; i < 4; ++i) acc = fmaf(a1[i], b1[i], acc);
#pragma unroll
  for (int m = 8; m >= 1; m >>= 1) acc += __shfl_xor(acc, m, 16);
  if (l16 == 0) {
    float denom = fmaxf(n1[src] * n2[dst], 1e-8f);
    // written once, never re-read: don't displace G rows from L2
    __builtin_nontemporal_store((acc / denom + 1.0f) * 0.5f, &out[e]);
  }
}

extern "C" void kernel_launch(void* const* d_in, const int* in_sizes, int n_in,
                              void* d_out, int out_size, void* d_ws, size_t ws_size,
                              hipStream_t stream) {
  const float* x     = (const float*)d_in[0];
  const int*   ei    = (const int*)d_in[1];
  const float* W1    = (const float*)d_in[2];
  const float* b1    = (const float*)d_in[3];
  const float* gamma = (const float*)d_in[4];
  const float* beta  = (const float*)d_in[5];
  const float* W2    = (const float*)d_in[6];
  const float* b2    = (const float*)d_in[7];
  float* out = (float*)d_out;

  char* ws = (char*)d_ws;
  bf16_t* H     = (bf16_t*)(ws + 0);            // 100000*128*2  = 25,600,000
  float*  G1    = (float*)(ws + 25600000);      // 100000*128*4  = 51,200,000
  float*  G2    = (float*)(ws + 76800000);      // 51,200,000
  float*  n1    = (float*)(ws + 128000000);     // 400,000
  float*  n2    = (float*)(ws + 128400000);     // 400,000
  int*    c1    = (int*)(ws + 128800000);       // 400,000 (c2 contiguous after)
  int*    c2    = (int*)(ws + 129200000);       // 400,000
  float*  stats = (float*)(ws + 129600000);     // 512*4
  bf16_t* W2p1  = (bf16_t*)(ws + 129602048);    // 32,768
  bf16_t* W2p2  = (bf16_t*)(ws + 129634816);    // 32,768
  float*  b2p1  = (float*)(ws + 129667584);     // 512
  float*  b2p2  = (float*)(ws + 129668096);     // 512

  pae_zero<<<782, 256, 0, stream>>>(c1, stats);
  pae_count<<<(N_EDGES_C + 255) / 256, 256, 0, stream>>>(ei, c1, c2);
  pae_gemm1<<<(N_NODES_C + 63) / 64, 256, 0, stream>>>(x, W1, b1, H, N_NODES_C);
  pae_stats<<<(N_NODES_C + 127) / 128, 256, 0, stream>>>(H, c1, c2, stats, N_NODES_C);
  pae_finalize<<<1, 256, 0, stream>>>(stats, gamma, beta, W2, b2, W2p1, W2p2, b2p1, b2p2);
  pae_gemm2<<<(N_NODES_C + 63) / 64, 256, 0, stream>>>(H, W2p1, b2p1, G1, N_NODES_C);
  pae_gemm2<<<(N_NODES_C + 63) / 64, 256, 0, stream>>>(H, W2p2, b2p2, G2, N_NODES_C);
  pae_norm<<<(N_NODES_C * 32 + 255) / 256, 256, 0, stream>>>(G1, G2, n1, n2, N_NODES_C);
  pae_edge<<<(N_EDGES_C * 16 + 255) / 256, 256, 0, stream>>>(ei, G1, G2, n1, n2, out);
}
